// VectorQuantizer_81149112090840
// MI455X (gfx1250) — matrix operand order repaired and checked by reading, not compiled, not run on hardware
//
#include <hip/hip_runtime.h>

typedef __attribute__((ext_vector_type(16))) __bf16 v16bf;
typedef __attribute__((ext_vector_type(8)))  __bf16 v8bf;
typedef __attribute__((ext_vector_type(8)))  float  v8f;
typedef int v4i __attribute__((vector_size(16)));

#define D_DIM   256
#define K_CODES 1024
#define HW      1024      // 32*32
#define N_VEC   32768     // 32*32*32
#define N_ELEM  8388608   // 32*256*32*32

// 16 stages: stage s = (d_step = s>>1, code_half = s&1); each 512 codes x 32 d
#define STAGE_ELEMS 16384  // 512*32 bf16 = 32 KB
#define N_STAGES    16

#if defined(__has_builtin)
#if __has_builtin(__builtin_amdgcn_global_load_async_to_lds_b128)
#define HAVE_ASYNC_LDS 1
#endif
#endif
#ifndef HAVE_ASYNC_LDS
#define HAVE_ASYNC_LDS 0
#endif

#if defined(__has_builtin)
#if __has_builtin(__builtin_amdgcn_s_wait_asynccnt)
#define WAIT_ASYNC(n) __builtin_amdgcn_s_wait_asynccnt(n)
#endif
#endif
#ifndef WAIT_ASYNC
#define WAIT_ASYNC(n) asm volatile("s_wait_asynccnt %0" ::"i"(n))
#endif

typedef const __attribute__((address_space(1))) char* gptr_t;
typedef __attribute__((address_space(3))) char*       lptr_t;
typedef __attribute__((address_space(1))) v4i*        g128_t;  // builtin src type
typedef __attribute__((address_space(3))) v4i*        l128_t;  // builtin dst type

// float -> bf16 round-to-nearest-even (bit trick)
static __device__ __forceinline__ __bf16 f2bf(float f) {
    unsigned u = __float_as_uint(f);
    unsigned r = (u + 0x7fffu + ((u >> 16) & 1u)) >> 16;
    unsigned short s = (unsigned short)r;
    return __builtin_bit_cast(__bf16, s);
}

// ---------------------------------------------------------------------------
// Kernel 1: codebook fp32 -> staged bf16 layout + ||e_k||^2
// ebf_s[stage][k&511][d&31], stage = (d>>5)*2 + (k>>9)
// ---------------------------------------------------------------------------
__global__ __launch_bounds__(256) void vq_prep_emb(
    const float* __restrict__ emb, __bf16* __restrict__ ebf_s,
    float* __restrict__ e2) {
    int k = blockIdx.x, d = threadIdx.x;
    float v = emb[k * D_DIM + d];
    int stage = (d >> 5) * 2 + (k >> 9);
    ebf_s[stage * STAGE_ELEMS + (k & 511) * 32 + (d & 31)] = f2bf(v);
    __shared__ float red[256];
    red[d] = v * v;
    __syncthreads();
    for (int o = 128; o > 0; o >>= 1) {
        if (d < o) red[d] += red[d + o];
        __syncthreads();
    }
    if (d == 0) e2[k] = red[0];
}

// ---------------------------------------------------------------------------
// Kernel 2: z [B,D,H,W] fp32 -> zbf [N,D] bf16 (N = b*1024 + h*32 + w)
// LDS transpose of 32(d) x 32(w) tiles. grid = (8 dtiles, 32 h, 32 b)
// ---------------------------------------------------------------------------
__global__ __launch_bounds__(256) void vq_prep_z(
    const float* __restrict__ z, __bf16* __restrict__ zbf) {
    int dt = blockIdx.x, h = blockIdx.y, b = blockIdx.z;
    int tx = threadIdx.x & 31, ty = threadIdx.x >> 5;
    __shared__ float tile[32][33];
#pragma unroll
    for (int j = 0; j < 4; ++j) {
        int d = dt * 32 + ty + j * 8;
        tile[ty + j * 8][tx] = z[((b * D_DIM + d) * 32 + h) * 32 + tx];  // tx = w
    }
    __syncthreads();
#pragma unroll
    for (int j = 0; j < 4; ++j) {
        int w = ty + j * 8;
        zbf[(b * HW + h * 32 + w) * D_DIM + dt * 32 + tx] = f2bf(tile[tx][w]);
    }
}

// ---------------------------------------------------------------------------
// Kernel 3: score GEMM + fused argmin. Block = 256 thr = 8 waves, 16 rows,
// all 1024 cols. B tiles double-buffered in LDS via async global->LDS copies.
// Wave w, phase h: cols h*512 + w*64 + j*16, j = 0..3.
// ---------------------------------------------------------------------------
__global__ __launch_bounds__(256) void vq_scores(
    const __bf16* __restrict__ zbf, const __bf16* __restrict__ ebf_s,
    const float* __restrict__ e2, int* __restrict__ idx_out) {
    __shared__ __align__(16) __bf16 bbuf[2][STAGE_ELEMS];  // 2 x 32 KB
    __shared__ float sval[16][8];
    __shared__ int   sidx[16][8];

    const int tid  = threadIdx.x;
    const int lane = tid & 31;
    const int wave = tid >> 5;  // 0..7
    const int row0 = blockIdx.x * 16;

    // A fragment (16x32 bf16 row-major): lane holds row lane&15,
    // elements 0..7 at K = akb..akb+7, elements 8..15 at K = 16+akb..
    const int arow = row0 + (lane & 15);
    const int akb  = (lane >> 4) * 8;
    // B fragment (32x16 bf16): lane holds col lane&15, same K order as A
    const int bcol = lane & 15;
    const int bkb  = (lane >> 4) * 8;

    v8f zero = {};
    v8f acc[2][4];
#pragma unroll
    for (int h = 0; h < 2; ++h)
#pragma unroll
        for (int j = 0; j < 4; ++j) acc[h][j] = zero;

#if HAVE_ASYNC_LDS
    // prime: stage 0 -> buffer 0 (8 x 16B chunks per thread, linear)
    {
        gptr_t g = (gptr_t)((const char*)ebf_s + tid * 16);
        lptr_t l = (lptr_t)((char*)&bbuf[0][0] + tid * 16);
#pragma unroll
        for (int i = 0; i < 8; ++i)
            __builtin_amdgcn_global_load_async_to_lds_b128(
                (g128_t)(g + i * 4096), (l128_t)(l + i * 4096), 0, 0);
    }
#endif

    for (int db8 = 0; db8 < 8; ++db8) {
        v16bf A;
#pragma unroll
        for (int h = 0; h < 2; ++h) {
            const int s = db8 * 2 + h;
#if HAVE_ASYNC_LDS
            if (s < N_STAGES - 1) {
                // issue stage s+1 into buffer h^1
                gptr_t g = (gptr_t)((const char*)(ebf_s + (size_t)(s + 1) * STAGE_ELEMS) + tid * 16);
                lptr_t l = (lptr_t)((char*)&bbuf[h ^ 1][0] + tid * 16);
#pragma unroll
                for (int i = 0; i < 8; ++i)
                    __builtin_amdgcn_global_load_async_to_lds_b128(
                        (g128_t)(g + i * 4096), (l128_t)(l + i * 4096), 0, 0);
                WAIT_ASYNC(8);   // async loads complete in order -> stage s done
            } else {
                WAIT_ASYNC(0);
            }
#else
            // synchronous staging fallback
            {
                const uint4* src = (const uint4*)(ebf_s + (size_t)s * STAGE_ELEMS);
                uint4* dst = (uint4*)&bbuf[h][0];
#pragma unroll
                for (int i = 0; i < 8; ++i) dst[tid + 256 * i] = src[tid + 256 * i];
            }
#endif
            __syncthreads();

            if (h == 0) {  // A fragment shared by both code-halves of this d-step
                const __bf16* ap = zbf + arow * D_DIM + db8 * 32 + akb;
                v8bf alo = *(const v8bf*)(ap);
                v8bf ahi = *(const v8bf*)(ap + 16);
                A = __builtin_shufflevector(alo, ahi, 0, 1, 2, 3, 4, 5, 6, 7,
                                            8, 9, 10, 11, 12, 13, 14, 15);
            }

            const __bf16* bb = &bbuf[h][0];
#pragma unroll
            for (int j = 0; j < 4; ++j) {
                int c = wave * 64 + j * 16 + bcol;       // code-local in half
                const __bf16* bp = bb + c * 32 + bkb;
                v8bf blo = *(const v8bf*)(bp);
                v8bf bhi = *(const v8bf*)(bp + 16);
                v16bf B = __builtin_shufflevector(blo, bhi, 0, 1, 2, 3, 4, 5, 6,
                                                  7, 8, 9, 10, 11, 12, 13, 14, 15);
                acc[h][j] = __builtin_amdgcn_wmma_f32_16x16x32_bf16(
                    false, A, false, B, (short)0, acc[h][j], false, false);
            }
            __syncthreads();  // all waves done reading before next stage overwrites
        }
    }

    // ---- fused argmin: score = e2[col] - 2*dot ----
    // C/D layout: lane L holds col N = L&15; VGPR v holds row (L>>4)*8 + v.
    float best[8];
    int   bidx[8];
#pragma unroll
    for (int v = 0; v < 8; ++v) { best[v] = __FLT_MAX__; bidx[v] = 0x7fffffff; }
#pragma unroll
    for (int h = 0; h < 2; ++h) {
#pragma unroll
        for (int j = 0; j < 4; ++j) {
            int   col = h * 512 + wave * 64 + j * 16 + (lane & 15);
            float e2c = e2[col];
#pragma unroll
            for (int v = 0; v < 8; ++v) {
                float s = e2c - 2.0f * acc[h][j][v];
                if (s < best[v]) { best[v] = s; bidx[v] = col; }
            }
        }
    }
#pragma unroll
    for (int off = 1; off <= 8; off <<= 1) {
#pragma unroll
        for (int v = 0; v < 8; ++v) {
            float ov = __shfl_xor(best[v], off, 32);
            int   oi = __shfl_xor(bidx[v], off, 32);
            if (ov < best[v] || (ov == best[v] && oi < bidx[v])) {
                best[v] = ov; bidx[v] = oi;
            }
        }
    }
    if ((lane & 15) == 0) {
        int rbase = (lane >> 4) * 8;
#pragma unroll
        for (int v = 0; v < 8; ++v) {
            sval[rbase + v][wave] = best[v];
            sidx[rbase + v][wave] = bidx[v];
        }
    }
    __syncthreads();
    if (tid < 16) {
        int r = tid;
        float bv = sval[r][0];
        int   bi = sidx[r][0];
#pragma unroll
        for (int wv = 1; wv < 8; ++wv) {
            float v = sval[r][wv];
            int   i = sidx[r][wv];
            if (v < bv || (v == bv && i < bi)) { bv = v; bi = i; }
        }
        idx_out[row0 + r] = bi;
    }
}

// ---------------------------------------------------------------------------
// Kernel 4: gather + output (fp32 exact) + per-block partial SSE.
// ---------------------------------------------------------------------------
__global__ __launch_bounds__(256) void vq_gather(
    const float* __restrict__ z, const float* __restrict__ emb,
    const int* __restrict__ idx, float* __restrict__ out,
    float* __restrict__ partial) {
    int tid = blockIdx.x * 256 + threadIdx.x;
    int hw  = tid & 1023;
    int t2  = tid >> 10;
    int d   = t2 & 255;
    int b   = t2 >> 8;
    int n   = b * HW + hw;
    int k   = idx[n];
    float e  = emb[k * D_DIM + d];
    float zv = z[tid];
    out[tid] = e;  // quantized_st == quantized numerically
    float diff = e - zv;
    __shared__ float red[256];
    red[threadIdx.x] = diff * diff;
    __syncthreads();
    for (int o = 128; o > 0; o >>= 1) {
        if (threadIdx.x < o) red[threadIdx.x] += red[threadIdx.x + o];
        __syncthreads();
    }
    if (threadIdx.x == 0) partial[blockIdx.x] = red[0];
}

// ---------------------------------------------------------------------------
// Kernel 5: deterministic final reduction; loss = 1.25 * mse
// ---------------------------------------------------------------------------
__global__ __launch_bounds__(256) void vq_loss(
    const float* __restrict__ partial, float* __restrict__ loss_out) {
    float s = 0.0f;
    for (int i = 0; i < N_VEC / 256; ++i)
        s += partial[threadIdx.x + 256 * i];
    __shared__ float red[256];
    red[threadIdx.x] = s;
    __syncthreads();
    for (int o = 128; o > 0; o >>= 1) {
        if (threadIdx.x < o) red[threadIdx.x] += red[threadIdx.x + o];
        __syncthreads();
    }
    if (threadIdx.x == 0)
        loss_out[0] = 1.25f * (red[0] / (float)N_ELEM);
}

extern "C" void kernel_launch(void* const* d_in, const int* in_sizes, int n_in,
                              void* d_out, int out_size, void* d_ws, size_t ws_size,
                              hipStream_t stream) {
    (void)in_sizes; (void)n_in; (void)out_size; (void)ws_size;
    const float* z   = (const float*)d_in[0];   // [32,256,32,32]
    const float* emb = (const float*)d_in[1];   // [1024,256]
    float* out = (float*)d_out;                 // 8388608 + 1

    char* ws = (char*)d_ws;
    __bf16* zbf   = (__bf16*)(ws);                                     // 16 MB
    __bf16* ebf_s = (__bf16*)(ws + 16777216);                          // 512 KB
    float*  e2    = (float*) (ws + 16777216 + 524288);                 // 4 KB
    int*    idx   = (int*)   (ws + 16777216 + 524288 + 4096);          // 128 KB
    float*  part  = (float*) (ws + 16777216 + 524288 + 4096 + 131072); // 128 KB

    vq_prep_emb<<<K_CODES, 256, 0, stream>>>(emb, ebf_s, e2);
    dim3 gz(8, 32, 32);
    vq_prep_z<<<gz, 256, 0, stream>>>(z, zbf);
    vq_scores<<<N_VEC / 16, 256, 0, stream>>>(zbf, ebf_s, e2, idx);
    vq_gather<<<N_ELEM / 256, 256, 0, stream>>>(z, emb, idx, out, part);
    vq_loss<<<1, 256, 0, stream>>>(part, out + N_ELEM);
}
